// GraphLearn_54906861912286
// MI455X (gfx1250) — matrix operand order repaired and checked by reading, not compiled, not run on hardware
//
#include <hip/hip_runtime.h>

typedef __attribute__((ext_vector_type(16))) _Float16 v16h;
typedef __attribute__((ext_vector_type(8)))  float    v8f;
typedef __attribute__((ext_vector_type(4)))  int      v4i;
typedef __attribute__((address_space(3)))    v4i      lds_v4i;

#define NN   8192
#define DD   256
#define BLK  128
#define LDA  264   // 256 halves + 8 pad -> 132 dwords/row, bank-conflict-free frags

// Async global->LDS staging path (CDNA5 GLOBAL_LOAD_ASYNC_TO_LDS_B128).
#if defined(__gfx1250__) && \
    __has_builtin(__builtin_amdgcn_global_load_async_to_lds_b128) && \
    __has_builtin(__builtin_amdgcn_s_wait_asynccnt)
#define HAVE_ASYNC_LDS 1
#else
#define HAVE_ASYNC_LDS 0
#endif

// ---------------------------------------------------------------------------
// Prep: x (f32, row-major) -> xh (f16, row-major) and exact f32 row sq-norms.
// One wave per row: 32 lanes x 8 elements.
// ---------------------------------------------------------------------------
__global__ __launch_bounds__(256) void prep_kernel(const float* __restrict__ x,
                                                   _Float16* __restrict__ xh,
                                                   float* __restrict__ sq) {
    const int row  = blockIdx.x * blockDim.y + threadIdx.y;
    const int lane = threadIdx.x;
    const float4* xr = reinterpret_cast<const float4*>(x + (size_t)row * DD);
    float4 v0 = xr[lane * 2 + 0];
    float4 v1 = xr[lane * 2 + 1];
    float s = v0.x * v0.x + v0.y * v0.y + v0.z * v0.z + v0.w * v0.w
            + v1.x * v1.x + v1.y * v1.y + v1.z * v1.z + v1.w * v1.w;
#pragma unroll
    for (int off = 16; off > 0; off >>= 1) s += __shfl_xor(s, off, 32);
    _Float16 h[8] = {(_Float16)v0.x, (_Float16)v0.y, (_Float16)v0.z, (_Float16)v0.w,
                     (_Float16)v1.x, (_Float16)v1.y, (_Float16)v1.z, (_Float16)v1.w};
    *reinterpret_cast<int4*>(xh + (size_t)row * DD + lane * 8) =
        *reinterpret_cast<int4*>(h);
    if (lane == 0) sq[row] = s;
}

// ---------------------------------------------------------------------------
// Main: each block -> 128x128 output tile. A/B rows (full K=256, f16) staged
// in LDS once (async DMA path when available); 8 waves each own a 2x4 grid of
// 16x16 WMMA tiles (V_WMMA_F32_16X16X32_F16, f32 accumulate).
// ---------------------------------------------------------------------------
__global__ __launch_bounds__(256) void dist_kernel(const _Float16* __restrict__ xh,
                                                   const float* __restrict__ sq,
                                                   const float* __restrict__ thr_p,
                                                   const float* __restrict__ t_p,
                                                   float* __restrict__ out) {
    extern __shared__ __align__(16) char smem_raw[];
    _Float16* Ah  = reinterpret_cast<_Float16*>(smem_raw);
    _Float16* Bh  = Ah + BLK * LDA;
    float*    sqA = reinterpret_cast<float*>(Bh + BLK * LDA);
    float*    sqB = sqA + BLK;

    const int tid = threadIdx.x;
    const int ib  = blockIdx.y * BLK;
    const int jb  = blockIdx.x * BLK;

    // Stage 128 A-rows + 128 B-rows of f16 (512 B each) into LDS.
    for (int idx = tid; idx < BLK * 32; idx += 256) {
        const int r = idx >> 5, s = idx & 31;
        const _Float16* gA = xh + (size_t)(ib + r) * DD + s * 8;
        const _Float16* gB = xh + (size_t)(jb + r) * DD + s * 8;
        _Float16* lA = Ah + r * LDA + s * 8;
        _Float16* lB = Bh + r * LDA + s * 8;
#if HAVE_ASYNC_LDS
        __builtin_amdgcn_global_load_async_to_lds_b128(
            (v4i*)gA, (lds_v4i*)lA, 0, 0);
        __builtin_amdgcn_global_load_async_to_lds_b128(
            (v4i*)gB, (lds_v4i*)lB, 0, 0);
#else
        *reinterpret_cast<int4*>(lA) = *reinterpret_cast<const int4*>(gA);
        *reinterpret_cast<int4*>(lB) = *reinterpret_cast<const int4*>(gB);
#endif
    }
    if (tid < BLK) sqA[tid] = sq[ib + tid];
    else           sqB[tid - BLK] = sq[jb + tid - BLK];
#if HAVE_ASYNC_LDS
    __builtin_amdgcn_s_wait_asynccnt(0);
#endif
    __syncthreads();

    const int lane = tid & 31;
    const int w    = tid >> 5;
    const int g    = lane >> 4;   // half-wave: selects K sub-block
    const int l16  = lane & 15;
    const int ti0  = (w & 3) * 2; // 2 tile-rows per wave
    const int tj0  = (w >> 2) * 4;// 4 tile-cols per wave

    v8f acc[2][4];
#pragma unroll
    for (int i = 0; i < 2; ++i)
#pragma unroll
        for (int j = 0; j < 4; ++j) acc[i][j] = (v8f){};

    for (int kc = 0; kc < DD / 32; ++kc) {
        v16h a[2], b[4];
        // A frag (ISA 7.12.2): lane holds row l16, K = {g*8..g*8+7, 16+g*8..}
#pragma unroll
        for (int i = 0; i < 2; ++i) {
            const _Float16* p = Ah + ((ti0 + i) * 16 + l16) * LDA + kc * 32 + g * 8;
            reinterpret_cast<int4*>(&a[i])[0] = *reinterpret_cast<const int4*>(p);
            reinterpret_cast<int4*>(&a[i])[1] = *reinterpret_cast<const int4*>(p + 16);
        }
        // B frag: lane holds column l16 (a row of x), K = {g*8..g*8+7, 16+g*8..}
#pragma unroll
        for (int j = 0; j < 4; ++j) {
            const _Float16* p = Bh + ((tj0 + j) * 16 + l16) * LDA + kc * 32 + g * 8;
            reinterpret_cast<int4*>(&b[j])[0] = *reinterpret_cast<const int4*>(p);
            reinterpret_cast<int4*>(&b[j])[1] = *reinterpret_cast<const int4*>(p + 16);
        }
#pragma unroll
        for (int i = 0; i < 2; ++i)
#pragma unroll
            for (int j = 0; j < 4; ++j)
                acc[i][j] = __builtin_amdgcn_wmma_f32_16x16x32_f16(
                    false, a[i], false, b[j], (short)0, acc[i][j], false, false);
    }

    // Epilogue: d2 = |xi|^2 + |xj|^2 - 2*dot ; out = 1/(1+exp((sqrt+thr)*t)).
    // Output is streamed once (256 MB) -> non-temporal stores keep L2 for x.
    const float thr = thr_p[0];
    const float tt  = t_p[0];
#pragma unroll
    for (int i = 0; i < 2; ++i) {
#pragma unroll
        for (int j = 0; j < 4; ++j) {
            const int   n_loc = (tj0 + j) * 16 + l16;
            const float sqn   = sqB[n_loc];
            const size_t gj   = (size_t)(jb + n_loc);
#pragma unroll
            for (int r = 0; r < 8; ++r) {
                const int   m_loc = (ti0 + i) * 16 + g * 8 + r;
                const float d2    = sqA[m_loc] + sqn - 2.0f * acc[i][j][r];
                const float dist  = sqrtf(fmaxf(d2, 1e-12f));
                const float z     = (dist + thr) * tt;
                const float res   = 1.0f / (1.0f + __expf(z));
                __builtin_nontemporal_store(res, out + (size_t)(ib + m_loc) * NN + gj);
            }
        }
    }
}

extern "C" void kernel_launch(void* const* d_in, const int* in_sizes, int n_in,
                              void* d_out, int out_size, void* d_ws, size_t ws_size,
                              hipStream_t stream) {
    const float* x         = (const float*)d_in[0];
    const float* threshold = (const float*)d_in[1];
    const float* t         = (const float*)d_in[2];
    float*       out       = (float*)d_out;

    _Float16* xh = (_Float16*)d_ws;                                   // 4 MB
    float*    sq = (float*)((char*)d_ws + (size_t)NN * DD * sizeof(_Float16));

    prep_kernel<<<dim3(NN / 8), dim3(32, 8), 0, stream>>>(x, xh, sq);

    const size_t shmem = (size_t)(2 * BLK * LDA) * sizeof(_Float16)
                       + (size_t)(2 * BLK) * sizeof(float);           // ~133 KB
    (void)hipFuncSetAttribute((const void*)dist_kernel,
                              hipFuncAttributeMaxDynamicSharedMemorySize,
                              (int)shmem);
    dist_kernel<<<dim3(NN / BLK, NN / BLK), 256, shmem, stream>>>(
        xh, sq, threshold, t, out);
}
